// SimpleTreeSAT_31894427140700
// MI455X (gfx1250) — matrix operand order repaired and checked by reading, not compiled, not run on hardware
//
#include <hip/hip_runtime.h>

// ---------------- types ----------------
typedef __attribute__((ext_vector_type(8)))  __bf16 v8bf;
typedef __attribute__((ext_vector_type(16))) __bf16 v16bf;
typedef __attribute__((ext_vector_type(8)))  float  v8f;

#define BS   64
#define OPS  512
#define NN   1024
#define DD   64

__device__ inline v8f vzero8() {
  v8f z;
#pragma unroll
  for (int i = 0; i < 8; ++i) z[i] = 0.f;
  return z;
}

// A fragment 16x32 bf16 (M x K), row-major source, leading dim ld (elements).
// ISA layout: lanes 0-15 row m=lane, lanes16-31 same rows, K interleave:
// elem e<8 -> K = kk + e + 8*half ; e>=8 -> K = kk + (e+8) + 8*half
__device__ inline v16bf frag_a(const __bf16* base, int ld, int row0, int kk) {
  int lane = threadIdx.x & 31;
  int m = row0 + (lane & 15);
  int h = (lane >> 4) & 1;
  const __bf16* p = base + (size_t)m * ld + kk;
  v8bf lo = *(const v8bf*)(p + 8 * h);
  v8bf hi = *(const v8bf*)(p + 16 + 8 * h);
  v16bf r;
#pragma unroll
  for (int i = 0; i < 8; ++i) { r[i] = lo[i]; r[8 + i] = hi[i]; }
  return r;
}

// B fragment 32x16 bf16 (K x N) built from W[n][k] row-major (so B = W^T).
// ISA layout: lane n = lane&15, half = lane>>4; elem e<8 -> K = kk + e + 8*half ; e>=8 -> K = kk + (e+8) + 8*half
__device__ inline v16bf frag_b(const __bf16* W, int ld, int n0, int kk) {
  int lane = threadIdx.x & 31;
  int n = n0 + (lane & 15);
  int h = (lane >> 4) & 1;
  const __bf16* p = W + (size_t)n * ld + kk;
  v8bf lo = *(const v8bf*)(p + 8 * h);
  v8bf hi = *(const v8bf*)(p + 16 + 8 * h);
  v16bf r;
#pragma unroll
  for (int i = 0; i < 8; ++i) { r[i] = lo[i]; r[8 + i] = hi[i]; }
  return r;
}

#define WMMA(a, b, c) \
  __builtin_amdgcn_wmma_f32_16x16x32_bf16(false, (a), false, (b), (short)0, (c), false, false)

__device__ inline float sigm(float x) { return 1.f / (1.f + expf(-x)); }

// ---------------- prep kernels ----------------
__global__ void k_cvt(const float* __restrict__ s, __bf16* __restrict__ d, int n) {
  int i = blockIdx.x * 256 + threadIdx.x;
  if (i < n) d[i] = (__bf16)s[i];
}

__global__ void k_prep_adj(const int* __restrict__ mxs,
                           __bf16* __restrict__ A, __bf16* __restrict__ AT) {
  long long t = (long long)blockIdx.x * 256 + threadIdx.x;  // over BS*OPS*NN
  int b = (int)(t >> 19);
  int rem = (int)(t & ((1 << 19) - 1));
  int op = rem >> 10;
  int node = rem & 1023;
  __bf16 v = (__bf16)(float)mxs[t];
  A[t] = v;
  AT[(size_t)b * NN * OPS + (size_t)node * OPS + op] = v;
}

__global__ void k_prep_state(const int* __restrict__ cons, const int* __restrict__ negs,
                             const float* __restrict__ con_emb, const float* __restrict__ neg_emb,
                             const float* __restrict__ var_emb, const float* __restrict__ start_emb,
                             __bf16* __restrict__ vty, __bf16* __restrict__ vbuf,
                             __bf16* __restrict__ hbuf, float* __restrict__ cbuf) {
  long long t = (long long)blockIdx.x * 256 + threadIdx.x;  // over BS*NN*DD
  int b = (int)(t >> 16);
  int i = (int)((t >> 6) & 1023);
  int d = (int)(t & 63);
  float tv;
  if (i < OPS) {
    float c = (float)cons[b * OPS + i];
    float g = (float)negs[b * OPS + i];
    tv = c * con_emb[d] + g * neg_emb[d];
  } else {
    tv = var_emb[d];
  }
  vty[t] = (__bf16)tv;
  vbuf[t] = (__bf16)start_emb[d];
  hbuf[t] = (__bf16)0.0f;
  cbuf[t] = 0.0f;
}

// ---------------- step kernel A: message MLPs ----------------
// x = [v | vtypes] (128), hidden = relu(x@W1^T+b1) (128), msg = hidden@W2^T+b2 (64)
// outputs stored transposed: msgT[b][feat(64)][node(1024)]
__global__ __launch_bounds__(128) void k_msg(
    const __bf16* __restrict__ vbuf, const __bf16* __restrict__ vty,
    const __bf16* __restrict__ Wm11, const float* __restrict__ bm11,
    const __bf16* __restrict__ Wm12, const float* __restrict__ bm12,
    const __bf16* __restrict__ Wm21, const float* __restrict__ bm21,
    const __bf16* __restrict__ Wm22, const float* __restrict__ bm22,
    __bf16* __restrict__ m1T, __bf16* __restrict__ m2T) {
  __shared__ __bf16 lds[4 * 16 * 128];
  int wave = threadIdx.x >> 5;
  int lane = threadIdx.x & 31;
  int tile = blockIdx.x * 4 + wave;  // 0..4095 over BS*NN/16
  int row0 = tile * 16;
  __bf16* hl = &lds[wave * 16 * 128];
  int n = lane & 15, h = lane >> 4;

  for (int pass = 0; pass < 2; ++pass) {
    const __bf16* W1 = pass ? Wm21 : Wm11;
    const float*  b1 = pass ? bm21 : bm11;
    const __bf16* W2 = pass ? Wm22 : Wm12;
    const float*  b2 = pass ? bm22 : bm12;
    __bf16* mT = pass ? m2T : m1T;

    v8f acc[8];
#pragma unroll
    for (int i = 0; i < 8; ++i) acc[i] = vzero8();

    for (int kc = 0; kc < 4; ++kc) {
      int kk = kc * 32;
      const __bf16* src = (kk < 64) ? vbuf : vty;
      v16bf a = frag_a(src, DD, row0, kk & 63);
#pragma unroll
      for (int nt = 0; nt < 8; ++nt) {
        v16bf bb = frag_b(W1, 128, nt * 16, kk);
        acc[nt] = WMMA(a, bb, acc[nt]);
      }
    }
    // bias + relu -> per-wave LDS tile [16 x 128]
#pragma unroll
    for (int nt = 0; nt < 8; ++nt) {
      int col = nt * 16 + n;
      float bias = b1[col];
#pragma unroll
      for (int r = 0; r < 8; ++r) {
        float v = acc[nt][r] + bias;
        hl[(r + 8 * h) * 128 + col] = (__bf16)(v > 0.f ? v : 0.f);
      }
    }
    // stage 2
    v8f acc2[4];
#pragma unroll
    for (int i = 0; i < 4; ++i) acc2[i] = vzero8();
    for (int kc = 0; kc < 4; ++kc) {
      v16bf a = frag_a(hl, 128, 0, kc * 32);
#pragma unroll
      for (int nt = 0; nt < 4; ++nt) {
        v16bf bb = frag_b(W2, 128, nt * 16, kc * 32);
        acc2[nt] = WMMA(a, bb, acc2[nt]);
      }
    }
    int bidx = row0 >> 10;
    int node0 = row0 & 1023;
    __bf16* dst = mT + (size_t)bidx * DD * NN;
#pragma unroll
    for (int nt = 0; nt < 4; ++nt) {
      int col = nt * 16 + n;
      float bias = b2[col];
#pragma unroll
      for (int r = 0; r < 8; ++r)
        dst[(size_t)col * NN + node0 + r + 8 * h] = (__bf16)(acc2[nt][r] + bias);
    }
  }
}

// ---------------- step kernel B: s = Adj@msg1 + AdjT@msg2 ----------------
__global__ __launch_bounds__(128) void k_adj(
    const __bf16* __restrict__ Abf, const __bf16* __restrict__ ATbf,
    const __bf16* __restrict__ m1T, const __bf16* __restrict__ m2T,
    __bf16* __restrict__ sbuf) {
  int wave = threadIdx.x >> 5;
  int lane = threadIdx.x & 31;
  int tile = blockIdx.x * 4 + wave;  // 0..4095
  int b = tile >> 6;
  int mt = tile & 63;
  int node0 = mt * 16;
  const __bf16* A  = Abf  + (size_t)b * OPS * NN;
  const __bf16* AT = ATbf + (size_t)b * NN * OPS;
  const __bf16* M1 = m1T + (size_t)b * DD * NN;
  const __bf16* M2 = m2T + (size_t)b * DD * NN;

  v8f acc[4];
#pragma unroll
  for (int i = 0; i < 4; ++i) acc[i] = vzero8();

  if (mt < 32) {  // wave-uniform: only op rows have Adj contributions
    for (int kc = 0; kc < 32; ++kc) {
      v16bf a = frag_a(A, NN, node0, kc * 32);
#pragma unroll
      for (int nt = 0; nt < 4; ++nt) {
        v16bf bb = frag_b(M1, NN, nt * 16, kc * 32);
        acc[nt] = WMMA(a, bb, acc[nt]);
      }
    }
  }
  for (int kc = 0; kc < 16; ++kc) {  // K over op nodes (512)
    v16bf a = frag_a(AT, OPS, node0, kc * 32);
#pragma unroll
    for (int nt = 0; nt < 4; ++nt) {
      v16bf bb = frag_b(M2, NN, nt * 16, kc * 32);
      acc[nt] = WMMA(a, bb, acc[nt]);
    }
  }
  int n = lane & 15, h = lane >> 4;
  __bf16* dst = sbuf + (size_t)b * NN * DD;
#pragma unroll
  for (int nt = 0; nt < 4; ++nt) {
    int col = nt * 16 + n;
#pragma unroll
    for (int r = 0; r < 8; ++r)
      dst[(size_t)(node0 + r + 8 * h) * DD + col] = (__bf16)acc[nt][r];
  }
}

// ---------------- step kernel C: gates + LSTM update ----------------
__global__ __launch_bounds__(128) void k_lstm(
    const __bf16* __restrict__ sbuf, __bf16* __restrict__ hbuf,
    const __bf16* __restrict__ Wih, const __bf16* __restrict__ Whh,
    const float* __restrict__ bih, const float* __restrict__ bhh,
    float* __restrict__ cbuf, __bf16* __restrict__ vbuf) {
  int wave = threadIdx.x >> 5;
  int lane = threadIdx.x & 31;
  int tile = blockIdx.x * 4 + wave;
  int row0 = tile * 16;

  v8f acc[16];
#pragma unroll
  for (int i = 0; i < 16; ++i) acc[i] = vzero8();

  for (int kc = 0; kc < 2; ++kc) {
    v16bf a = frag_a(sbuf, DD, row0, kc * 32);
#pragma unroll
    for (int t = 0; t < 16; ++t) {
      v16bf bb = frag_b(Wih, DD, t * 16, kc * 32);
      acc[t] = WMMA(a, bb, acc[t]);
    }
  }
  for (int kc = 0; kc < 2; ++kc) {
    v16bf a = frag_a(hbuf, DD, row0, kc * 32);
#pragma unroll
    for (int t = 0; t < 16; ++t) {
      v16bf bb = frag_b(Whh, DD, t * 16, kc * 32);
      acc[t] = WMMA(a, bb, acc[t]);
    }
  }
  int n = lane & 15, h = lane >> 4;
#pragma unroll
  for (int nt = 0; nt < 4; ++nt) {
    int col = nt * 16 + n;
    float bi = bih[col] + bhh[col];
    float bf = bih[64 + col] + bhh[64 + col];
    float bg = bih[128 + col] + bhh[128 + col];
    float bo = bih[192 + col] + bhh[192 + col];
#pragma unroll
    for (int r = 0; r < 8; ++r) {
      size_t idx = (size_t)(row0 + r + 8 * h) * DD + col;
      float iv = sigm(acc[nt][r] + bi);
      float fv = sigm(acc[4 + nt][r] + bf);
      float gv = tanhf(acc[8 + nt][r] + bg);
      float ov = sigm(acc[12 + nt][r] + bo);
      float cn = fv * cbuf[idx] + iv * gv;
      float hn = ov * tanhf(cn);
      cbuf[idx] = cn;
      __bf16 hb = (__bf16)hn;
      vbuf[idx] = hb;
      hbuf[idx] = hb;
    }
  }
}

// ---------------- readout ----------------
__global__ void k_readout(const __bf16* __restrict__ vbuf,
                          const float* __restrict__ Wc1, const float* __restrict__ bc1,
                          const float* __restrict__ Wc2, float* __restrict__ out) {
  int b = threadIdx.x;
  if (b >= BS) return;
  float v0[DD];
#pragma unroll
  for (int d = 0; d < DD; ++d) v0[d] = (float)vbuf[(size_t)b * NN * DD + d];
  float s2 = 0.f;
  for (int j = 0; j < DD; ++j) {
    float a = bc1[j];
    for (int d = 0; d < DD; ++d) a += v0[d] * Wc1[j * DD + d];
    a = a > 0.f ? a : 0.f;
    s2 += a * Wc2[j];
  }
  out[b] = 1.f / (1.f + expf(-s2));
}

// ---------------- host launcher ----------------
extern "C" void kernel_launch(void* const* d_in, const int* in_sizes, int n_in,
                              void* d_out, int out_size, void* d_ws, size_t ws_size,
                              hipStream_t stream) {
  const int*   mxs       = (const int*)d_in[0];
  const int*   cons      = (const int*)d_in[1];
  const int*   negs      = (const int*)d_in[2];
  const float* start_emb = (const float*)d_in[3];
  const float* con_emb   = (const float*)d_in[4];
  const float* neg_emb   = (const float*)d_in[5];
  const float* var_emb   = (const float*)d_in[6];
  const float* Wm11 = (const float*)d_in[7];  const float* bm11 = (const float*)d_in[8];
  const float* Wm12 = (const float*)d_in[9];  const float* bm12 = (const float*)d_in[10];
  const float* Wm21 = (const float*)d_in[11]; const float* bm21 = (const float*)d_in[12];
  const float* Wm22 = (const float*)d_in[13]; const float* bm22 = (const float*)d_in[14];
  const float* Wih  = (const float*)d_in[15]; const float* Whh  = (const float*)d_in[16];
  const float* bih  = (const float*)d_in[17]; const float* bhh  = (const float*)d_in[18];
  const float* Wc1  = (const float*)d_in[19]; const float* bc1  = (const float*)d_in[20];
  const float* Wc2  = (const float*)d_in[21];
  float* out = (float*)d_out;

  size_t off = 0;
  auto carve = [&](size_t bytes) {
    void* p = (char*)d_ws + off;
    off += (bytes + 255) & ~(size_t)255;
    return p;
  };
  __bf16* Abf  = (__bf16*)carve((size_t)BS * OPS * NN * 2);
  __bf16* ATbf = (__bf16*)carve((size_t)BS * NN * OPS * 2);
  __bf16* vty  = (__bf16*)carve((size_t)BS * NN * DD * 2);
  __bf16* vbuf = (__bf16*)carve((size_t)BS * NN * DD * 2);
  __bf16* hbuf = (__bf16*)carve((size_t)BS * NN * DD * 2);
  float*  cbuf = (float*) carve((size_t)BS * NN * DD * 4);
  __bf16* m1T  = (__bf16*)carve((size_t)BS * DD * NN * 2);
  __bf16* m2T  = (__bf16*)carve((size_t)BS * DD * NN * 2);
  __bf16* sbuf = (__bf16*)carve((size_t)BS * NN * DD * 2);
  __bf16* Wm11b = (__bf16*)carve(128 * 128 * 2);
  __bf16* Wm12b = (__bf16*)carve(64 * 128 * 2);
  __bf16* Wm21b = (__bf16*)carve(128 * 128 * 2);
  __bf16* Wm22b = (__bf16*)carve(64 * 128 * 2);
  __bf16* Wihb  = (__bf16*)carve(256 * 64 * 2);
  __bf16* Whhb  = (__bf16*)carve(256 * 64 * 2);
  (void)ws_size; (void)n_in; (void)in_sizes; (void)out_size;

  // weight conversions
  k_cvt<<<(16384 + 255) / 256, 256, 0, stream>>>(Wm11, Wm11b, 16384);
  k_cvt<<<(8192  + 255) / 256, 256, 0, stream>>>(Wm12, Wm12b, 8192);
  k_cvt<<<(16384 + 255) / 256, 256, 0, stream>>>(Wm21, Wm21b, 16384);
  k_cvt<<<(8192  + 255) / 256, 256, 0, stream>>>(Wm22, Wm22b, 8192);
  k_cvt<<<(16384 + 255) / 256, 256, 0, stream>>>(Wih, Wihb, 16384);
  k_cvt<<<(16384 + 255) / 256, 256, 0, stream>>>(Whh, Whhb, 16384);

  // adjacency + state init
  k_prep_adj<<<(BS * OPS * NN) / 256, 256, 0, stream>>>(mxs, Abf, ATbf);
  k_prep_state<<<(BS * NN * DD) / 256, 256, 0, stream>>>(
      cons, negs, con_emb, neg_emb, var_emb, start_emb, vty, vbuf, hbuf, cbuf);

  const int tiles = BS * NN / 16;          // 4096 waves
  const int blocks = tiles / 4;            // 4 waves per 128-thread block

  for (int step = 0; step < 16; ++step) {
    k_msg<<<blocks, 128, 0, stream>>>(vbuf, vty, Wm11b, bm11, Wm12b, bm12,
                                      Wm21b, bm21, Wm22b, bm22, m1T, m2T);
    k_adj<<<blocks, 128, 0, stream>>>(Abf, ATbf, m1T, m2T, sbuf);
    k_lstm<<<blocks, 128, 0, stream>>>(sbuf, hbuf, Wihb, Whhb, bih, bhh, cbuf, vbuf);
  }
  k_readout<<<1, 64, 0, stream>>>(vbuf, Wc1, bc1, Wc2, out);
}
